// SynthesizerAttention_73323681677941
// MI455X (gfx1250) — matrix operand order repaired and checked by reading, not compiled, not run on hardware
//
#include <hip/hip_runtime.h>
#include <hip/hip_bf16.h>

// ---------------------------------------------------------------------------
// CDNA5 (gfx1250) WMMA implementation of Synthesizer attention.
//   B=2, T=2048, D=1024, H=16, Dh=64.  All matmuls on v_wmma_f32_16x16x32_f16.
// Data movement:
//   * one-shot f32->f16 conversion of x / W (halves repeated L2 traffic)
//   * GEMMs: double-buffered GLOBAL_LOAD_ASYNC_TO_LDS_B128 (ASYNCcnt pipeline)
//   * attention: K_pos / V^T 64x64 tiles DMA'd once per workgroup via the
//     Tensor Data Mover (TENSOR_LOAD_TO_LDS, TENSORcnt double-buffer)
// Fragment layouts follow CDNA5 ISA 7.12.2 (wave32):
//   A 16x32 f16 : lane l (m=l%16, h=l/16) holds K = h*8+0..7 and 16+h*8+0..7
//   B 32x16 f16 : lane l (n=l%16, h=l/16) holds K = h*8+0..7 and 16+h*8+0..7 (col n)
//   C/D 16x16 f32: lane l (n=l%16) VGPR r holds M = r + 8*(l/16)
// ---------------------------------------------------------------------------

typedef _Float16 v16h __attribute__((ext_vector_type(16)));
typedef _Float16 v8h  __attribute__((ext_vector_type(8)));
typedef _Float16 v4h  __attribute__((ext_vector_type(4)));
typedef float    v8f  __attribute__((ext_vector_type(8)));

typedef int v4i __attribute__((vector_size(16)));
typedef __attribute__((address_space(1))) v4i* gv4ip;        // global int4*
typedef __attribute__((address_space(3))) v4i* lv4ip;        // LDS int4*
typedef __attribute__((address_space(3))) _Float16* lhp;     // LDS half*

#if __has_builtin(__builtin_amdgcn_global_load_async_to_lds_b128)
#define ATHENA_HAS_ASYNC 1
#else
#define ATHENA_HAS_ASYNC 0
#endif

#if __has_builtin(__builtin_amdgcn_tensor_load_to_lds)
#define ATHENA_HAS_TDM 1
#else
#define ATHENA_HAS_TDM 0
#endif

// Copy 16 bytes global -> LDS.  Async (ASYNCcnt) when the builtin exists.
__device__ __forceinline__ void copy16_g2l(const _Float16* g, _Float16* l) {
#if ATHENA_HAS_ASYNC
  __builtin_amdgcn_global_load_async_to_lds_b128((gv4ip)g, (lv4ip)l, 0, 0);
#else
  *(v8h*)l = *(const v8h*)g;
#endif
}

template <int N>
__device__ __forceinline__ void wait_async() {
#if ATHENA_HAS_ASYNC
#if __has_builtin(__builtin_amdgcn_s_wait_asynccnt)
  __builtin_amdgcn_s_wait_asynccnt(N);
#else
  asm volatile("s_wait_asynccnt %0" ::"n"(N) : "memory");
#endif
#endif
}

template <int N>
__device__ __forceinline__ void wait_tensor() {
#if ATHENA_HAS_TDM
#if __has_builtin(__builtin_amdgcn_s_wait_tensorcnt)
  __builtin_amdgcn_s_wait_tensorcnt(N);
#else
  asm volatile("s_wait_tensorcnt %0" ::"n"(N) : "memory");
#endif
#endif
}

#if ATHENA_HAS_TDM
// TDM 2D tile load: tile_h rows x 64 halves, row stride `stride` halves,
// packed contiguously into LDS.  D# per CDNA5 ISA 8.3/8.4 (2D: groups 2/3 = 0).
__device__ __forceinline__ void tdm_load_2d(const _Float16* g, _Float16* l,
                                            unsigned tile_h, unsigned stride) {
  typedef unsigned int u32x4 __attribute__((ext_vector_type(4)));
  typedef int i32x8 __attribute__((ext_vector_type(8)));
  typedef int i32x4 __attribute__((ext_vector_type(4)));
  const unsigned lds = (unsigned)(unsigned long long)(lhp)l;
  const unsigned long long ga = (unsigned long long)g;
  // group0: count=1 | lds_addr | global_addr[56:0] | type=2 ("image")
  u32x4 g0 = {1u, lds, (unsigned)ga,
              (unsigned)((ga >> 32) & 0x01FFFFFFu) | (2u << 30)};
  // group1: data_size=2B; tensor_dim0=64; tensor_dim1=tile_h;
  //         tile_dim0=64; tile_dim1=tile_h; tile_dim2=0; dim0_stride=stride
  i32x8 g1 = {(int)0x00010000u,
              (int)(64u << 16),
              (int)(tile_h << 16),
              (int)(64u << 16),
              (int)tile_h,
              (int)stride,
              0, 0};
  i32x4 z4 = {0, 0, 0, 0};
#if __clang_major__ >= 23
  i32x8 z8 = {0, 0, 0, 0, 0, 0, 0, 0};
  __builtin_amdgcn_tensor_load_to_lds(g0, g1, z4, z4, z8, 0);
#else
  __builtin_amdgcn_tensor_load_to_lds(g0, g1, z4, z4, 0);
#endif
}
#endif

__device__ __forceinline__ v16h cat8(v8h lo, v8h hi) {
  return __builtin_shufflevector(lo, hi, 0,1,2,3,4,5,6,7,8,9,10,11,12,13,14,15);
}

// A-fragment (16x32, M x K) from row-major f16, row stride rs (in halves).
__device__ __forceinline__ v16h load_a_frag(const _Float16* src, int rs) {
  const int lane = threadIdx.x & 31;
  const int m = lane & 15, half = lane >> 4;
  const _Float16* p = src + m * rs + half * 8;
  v8h lo = *(const v8h*)(p);        // K = half*8 + 0..7
  v8h hi = *(const v8h*)(p + 16);   // K = 16 + half*8 + 0..7
  return cat8(lo, hi);
}

// B-fragment (32x16, K x N) from an n-major ("Bt") buffer: Bt[n][k], stride rs.
__device__ __forceinline__ v16h load_bt_frag(const _Float16* src, int rs) {
  const int lane = threadIdx.x & 31;
  const int n = lane & 15, half = lane >> 4;
  const _Float16* p = src + n * rs + half * 8;  // K = half*8 + 0..7, 16 + half*8 + 0..7
  v8h lo = *(const v8h*)(p);
  v8h hi = *(const v8h*)(p + 16);
  return cat8(lo, hi);
}

__device__ __forceinline__ v8f wmma16(v16h a, v16h b, v8f c) {
  return __builtin_amdgcn_wmma_f32_16x16x32_f16(false, a, false, b,
                                                (short)0, c, false, false);
}

// ---------------------------------------------------------------------------
// One-shot f32 -> f16 conversion (vectorized 4-wide).
// ---------------------------------------------------------------------------
__global__ void cvt_f16_kernel(const float* __restrict__ src,
                               _Float16* __restrict__ dst, int n4) {
  int i = blockIdx.x * 256 + threadIdx.x;
  if (i >= n4) return;
  float4 v = *(const float4*)(src + (size_t)i * 4);
  v4h o = {(_Float16)v.x, (_Float16)v.y, (_Float16)v.z, (_Float16)v.w};
  *(v4h*)(dst + (size_t)i * 4) = o;
}

// ---------------------------------------------------------------------------
// k_pos[t][d] = cos(ang) + rotate_half(ones)*sin = cos -/+ sin  (f16)
// ---------------------------------------------------------------------------
__global__ void kpos_kernel(_Float16* __restrict__ Kp) {
  int i = blockIdx.x * 256 + threadIdx.x;
  if (i >= 2048 * 64) return;
  int t = i >> 6, d = i & 63;
  float invf = __powf(10000.f, -(float)(d & 31) / 32.f);
  float s, c;
  __sincosf((float)t * invf, &s, &c);
  Kp[i] = (_Float16)((d < 32) ? (c - s) : (c + s));
}

// ---------------------------------------------------------------------------
// Q/V projection: [4096,1024] @ W^T, 64x64 tile per block (4 waves x 16x64).
// Double-buffered async LDS staging of f16 A/B tiles.
//   z==0 : W=Wq, fuse RoPE + 1/sqrt(Dh), write Q  as f16 [BH][T][Dh]
//   z==1 : W=Wv,                         write Vt as f16 [BH][Dh][T]
// ---------------------------------------------------------------------------
__global__ void proj_qv_kernel(const _Float16* __restrict__ xh,
                               const _Float16* __restrict__ Wqh,
                               const _Float16* __restrict__ Wvh,
                               _Float16* __restrict__ Q,
                               _Float16* __restrict__ Vt) {
  __shared__ __align__(16) _Float16 As[2][64 * 40];
  __shared__ __align__(16) _Float16 Bs[2][64 * 40];
  const int tid = threadIdx.x, wave = tid >> 5, lane = tid & 31;
  const int n = lane & 15, half = lane >> 4;
  const int Mb = blockIdx.x * 64;          // row tile in [0,4096)
  const int Nb = blockIdx.y * 64;          // col tile in [0,1024) == one head
  const bool isV = blockIdx.z != 0;
  const _Float16* W = isV ? Wvh : Wqh;

  // Stage one 64x32 A tile + 64x32 Bt tile (16-byte chunks, 4 async ops/thread)
  auto stage = [&](int buf, int kb) {
#pragma unroll
    for (int it = 0; it < 2; ++it) {
      const int c = tid * 2 + it;          // chunk 0..255
      const int r = c >> 2, q = (c & 3) * 8;
      copy16_g2l(xh + (size_t)(Mb + r) * 1024 + kb + q, &As[buf][r * 40 + q]);
      copy16_g2l(W + (size_t)(Nb + r) * 1024 + kb + q, &Bs[buf][r * 40 + q]);
    }
  };

  v8f acc[4] = {};
  stage(0, 0);
  for (int kb = 0; kb < 1024; kb += 32) {
    const int cur = (kb >> 5) & 1;
    if (kb + 32 < 1024) {
      stage(cur ^ 1, kb + 32);   // prefetch next tile into other buffer
      wait_async<4>();           // wait for *current* tile's 4 ops only
    } else {
      wait_async<0>();
    }
    __syncthreads();
    v16h a = load_a_frag(&As[cur][(wave * 16) * 40], 40);
#pragma unroll
    for (int f = 0; f < 4; ++f)
      acc[f] = wmma16(a, load_bt_frag(&Bs[cur][(f * 16) * 40], 40), acc[f]);
    __syncthreads();
  }

  const int row0 = Mb + wave * 16;
  const int h = Nb >> 6;
  const int b = row0 >> 11;                 // tile never crosses batch bound
  if (!isV) {
    // RoPE: columns dh and dh+32 are fragments f and f+2 in the SAME lane,
    // and share the same angle (d % 32).  Fuse attention scale 1/8.
#pragma unroll
    for (int fp = 0; fp < 2; ++fp) {
      const int dh = n + 16 * fp;           // < 32
      const float invf = __powf(10000.f, -(float)dh / 32.f);
#pragma unroll
      for (int r = 0; r < 8; ++r) {
        const int m = r + 8 * half;
        const int t = (row0 + m) & 2047;
        float s, c;
        __sincosf((float)t * invf, &s, &c);
        const float q0 = acc[fp][r], q2 = acc[fp + 2][r];
        const size_t base = ((size_t)(b * 16 + h) * 2048 + t) * 64;
        Q[base + dh]      = (_Float16)((q0 * c - q2 * s) * 0.125f);
        Q[base + dh + 32] = (_Float16)((q2 * c + q0 * s) * 0.125f);
      }
    }
  } else {
    // Transposed store: per lane the 8 accumulator rows are consecutive t.
#pragma unroll
    for (int f = 0; f < 4; ++f) {
      const int dh = n + 16 * f;
      const int tb = (row0 & 2047) + 8 * half;
      v8h pack;
#pragma unroll
      for (int r = 0; r < 8; ++r) pack[r] = (_Float16)acc[f][r];
      *(v8h*)(Vt + ((size_t)(b * 16 + h) * 64 + dh) * 2048 + tb) = pack;
    }
  }
}

// ---------------------------------------------------------------------------
// Flash attention: block = (b,h) x 64 rows; 4 waves x 16 rows; stream 32
// j-tiles of 64.  K_pos / V^T tiles are TDM-DMA'd into LDS once per block
// (double-buffered on TENSORcnt); P re-layouts C->A via per-wave LDS.
// ---------------------------------------------------------------------------
__global__ void attn_kernel(const _Float16* __restrict__ Q,
                            const _Float16* __restrict__ Kp,
                            const _Float16* __restrict__ Vt,
                            _Float16* __restrict__ Oa) {
  __shared__ __align__(16) _Float16 Pl[4][16 * 72];
#if ATHENA_HAS_TDM
  __shared__ __align__(16) _Float16 Ks[2][64 * 64];  // K_pos tile [j][d]
  __shared__ __align__(16) _Float16 Vs[2][64 * 64];  // V^T  tile [dh][j]
#endif
  const int tid = threadIdx.x, wave = tid >> 5, lane = tid & 31;
  const int n = lane & 15, half = lane >> 4;
  const int bh = blockIdx.y;
  const int t0 = blockIdx.x * 64 + wave * 16;

  const _Float16* qrow = Q + ((size_t)bh * 2048 + t0) * 64;
  const v16h qa0 = load_a_frag(qrow, 64);        // K = d 0..31 (in registers
  const v16h qa1 = load_a_frag(qrow + 32, 64);   //     for all 32 j-tiles)

  v8f oacc[4] = {};
  float mi[8], li[8];
#pragma unroll
  for (int r = 0; r < 8; ++r) { mi[r] = -1e30f; li[r] = 0.f; }
  _Float16* Pw = &Pl[wave][0];

#if ATHENA_HAS_TDM
  if (wave == 0) {                               // prime tile 0
    tdm_load_2d(Kp, &Ks[0][0], 64, 64);
    tdm_load_2d(Vt + (size_t)bh * 64 * 2048, &Vs[0][0], 64, 2048);
  }
#endif

  for (int jt = 0; jt < 32; ++jt) {
    const int jb = jt * 64;
    const int cur = jt & 1;
#if ATHENA_HAS_TDM
    if (wave == 0) {
      if (jt + 1 < 32) {                         // DMA next tile, other buffer
        tdm_load_2d(Kp + (size_t)(jb + 64) * 64, &Ks[cur ^ 1][0], 64, 64);
        tdm_load_2d(Vt + (size_t)bh * 64 * 2048 + jb + 64, &Vs[cur ^ 1][0],
                    64, 2048);
        wait_tensor<2>();                        // current tile's 2 DMAs done
      } else {
        wait_tensor<0>();
      }
    }
    __syncthreads();
#else
    if (jt + 1 < 32) {                           // cache prefetch of next tile
      __builtin_prefetch(Kp + (size_t)(jb + 64) * 64 + lane * 256, 0, 0);
      __builtin_prefetch(Vt + ((size_t)bh * 64 + lane) * 2048 + jb + 64, 0, 0);
      __builtin_prefetch(
          Vt + ((size_t)bh * 64 + 32 + lane) * 2048 + jb + 64, 0, 0);
    }
#endif
    v8f s[4] = {};
#pragma unroll
    for (int f = 0; f < 4; ++f) {                 // S = (Q*scale) @ Kpos^T
#if ATHENA_HAS_TDM
      const _Float16* kb = &Ks[cur][(f * 16) * 64];
      s[f] = wmma16(qa0, load_bt_frag(kb, 64), s[f]);
      s[f] = wmma16(qa1, load_bt_frag(kb + 32, 64), s[f]);
#else
      const _Float16* kb = Kp + (size_t)(jb + f * 16) * 64;
      s[f] = wmma16(qa0, load_bt_frag(kb, 64), s[f]);
      s[f] = wmma16(qa1, load_bt_frag(kb + 32, 64), s[f]);
#endif
    }
#pragma unroll
    for (int r = 0; r < 8; ++r) {                 // online softmax per row
      float mt = fmaxf(fmaxf(s[0][r], s[1][r]), fmaxf(s[2][r], s[3][r]));
      mt = fmaxf(mt, __shfl_xor(mt, 1, 32));
      mt = fmaxf(mt, __shfl_xor(mt, 2, 32));
      mt = fmaxf(mt, __shfl_xor(mt, 4, 32));
      mt = fmaxf(mt, __shfl_xor(mt, 8, 32));      // row max over 16 lanes
      const float mn = fmaxf(mi[r], mt);
      const float corr = __expf(mi[r] - mn);
      float rs = 0.f;
#pragma unroll
      for (int f = 0; f < 4; ++f) {
        const float p = __expf(s[f][r] - mn);
        s[f][r] = p;
        rs += p;
      }
      rs += __shfl_xor(rs, 1, 32);
      rs += __shfl_xor(rs, 2, 32);
      rs += __shfl_xor(rs, 4, 32);
      rs += __shfl_xor(rs, 8, 32);
      li[r] = li[r] * corr + rs;
      mi[r] = mn;
#pragma unroll
      for (int f = 0; f < 4; ++f) oacc[f][r] *= corr;
      const int m = r + 8 * half;                 // spill P (f16) to LDS
#pragma unroll
      for (int f = 0; f < 4; ++f) Pw[m * 72 + n + 16 * f] = (_Float16)s[f][r];
    }
    // O += P @ V  (same-wave DS write->read is in-order; no barrier needed)
#pragma unroll
    for (int ks = 0; ks < 2; ++ks) {
      const v16h pa = load_a_frag(Pw + ks * 32, 72);
#pragma unroll
      for (int f = 0; f < 4; ++f) {
#if ATHENA_HAS_TDM
        const _Float16* vb = &Vs[cur][(f * 16) * 64] + ks * 32;
        oacc[f] = wmma16(pa, load_bt_frag(vb, 64), oacc[f]);
#else
        const _Float16* vb =
            Vt + ((size_t)bh * 64 + f * 16) * 2048 + jb + ks * 32;
        oacc[f] = wmma16(pa, load_bt_frag(vb, 2048), oacc[f]);
#endif
      }
    }
#if ATHENA_HAS_TDM
    __syncthreads();   // all waves done reading buf[cur] before it is re-DMAed
#endif
  }
  const int b = bh >> 4, h = bh & 15;
#pragma unroll
  for (int r = 0; r < 8; ++r) {
    const float inv = 1.f / li[r];
    const int m = r + 8 * half;
    const size_t row = (size_t)b * 2048 + t0 + m;
#pragma unroll
    for (int f = 0; f < 4; ++f)
      Oa[row * 1024 + h * 64 + n + 16 * f] = (_Float16)(oacc[f][r] * inv);
  }
}

// ---------------------------------------------------------------------------
// Output projection: out = Oa(f16) @ Woh^T + bo   (fp32 out)
// A fragments straight from global (row-major f16); B double-buffered async.
// ---------------------------------------------------------------------------
__global__ void out_proj_kernel(const _Float16* __restrict__ A,
                                const _Float16* __restrict__ Woh,
                                const float* __restrict__ bo,
                                float* __restrict__ out) {
  __shared__ __align__(16) _Float16 Bs[2][64 * 40];
  const int tid = threadIdx.x, wave = tid >> 5, lane = tid & 31;
  const int n = lane & 15, half = lane >> 4;
  const int Mb = blockIdx.x * 64, Nb = blockIdx.y * 64;

  auto stageB = [&](int buf, int kb) {
#pragma unroll
    for (int it = 0; it < 2; ++it) {
      const int c = tid * 2 + it;
      const int r = c >> 2, q = (c & 3) * 8;
      copy16_g2l(Woh + (size_t)(Nb + r) * 1024 + kb + q,
                 &Bs[buf][r * 40 + q]);
    }
  };

  v8f acc[4] = {};
  stageB(0, 0);
  for (int kb = 0; kb < 1024; kb += 32) {
    const int cur = (kb >> 5) & 1;
    if (kb + 32 < 1024) {
      stageB(cur ^ 1, kb + 32);
      wait_async<2>();
    } else {
      wait_async<0>();
    }
    __syncthreads();
    v16h a = load_a_frag(A + (size_t)(Mb + wave * 16) * 1024 + kb, 1024);
#pragma unroll
    for (int f = 0; f < 4; ++f)
      acc[f] = wmma16(a, load_bt_frag(&Bs[cur][(f * 16) * 40], 40), acc[f]);
    __syncthreads();
  }
#pragma unroll
  for (int f = 0; f < 4; ++f) {
    const float bias = bo[Nb + n + 16 * f];
#pragma unroll
    for (int r = 0; r < 8; ++r) {
      const int m = r + 8 * half;
      out[(size_t)(Mb + m) * 1024 + Nb + n + 16 * f] = acc[f][r] + bias;
    }
  }
}

// ---------------------------------------------------------------------------
extern "C" void kernel_launch(void* const* d_in, const int* in_sizes, int n_in,
                              void* d_out, int out_size, void* d_ws,
                              size_t ws_size, hipStream_t stream) {
  (void)in_sizes; (void)n_in; (void)out_size; (void)ws_size;
  const float* x  = (const float*)d_in[0];
  const float* Wq = (const float*)d_in[1];
  const float* Wv = (const float*)d_in[2];
  const float* Wo = (const float*)d_in[3];
  const float* bo = (const float*)d_in[4];

  // f16 workspace: Q 8MB | Vt 8MB | Kp 256KB | Oa 8MB | xh 8MB | W*h 2MB x3
  _Float16* Q   = (_Float16*)d_ws;
  _Float16* Vt  = Q   + (size_t)32 * 2048 * 64;
  _Float16* Kp  = Vt  + (size_t)32 * 2048 * 64;
  _Float16* Oa  = Kp  + (size_t)2048 * 64;
  _Float16* xh  = Oa  + (size_t)4096 * 1024;
  _Float16* Wqh = xh  + (size_t)4096 * 1024;
  _Float16* Wvh = Wqh + (size_t)1024 * 1024;
  _Float16* Woh = Wvh + (size_t)1024 * 1024;

  cvt_f16_kernel<<<4096, 256, 0, stream>>>(x, xh, 4096 * 1024 / 4);
  cvt_f16_kernel<<<1024, 256, 0, stream>>>(Wq, Wqh, 1024 * 1024 / 4);
  cvt_f16_kernel<<<1024, 256, 0, stream>>>(Wv, Wvh, 1024 * 1024 / 4);
  cvt_f16_kernel<<<1024, 256, 0, stream>>>(Wo, Woh, 1024 * 1024 / 4);
  kpos_kernel<<<(2048 * 64 + 255) / 256, 256, 0, stream>>>(Kp);

  dim3 g1(64, 16, 2);
  proj_qv_kernel<<<g1, 128, 0, stream>>>(xh, Wqh, Wvh, Q, Vt);

  dim3 g2(32, 32);
  attn_kernel<<<g2, 128, 0, stream>>>(Q, Kp, Vt, Oa);

  dim3 g3(64, 16);
  out_proj_kernel<<<g3, 128, 0, stream>>>(Oa, Woh, bo, (float*)d_out);
}
